// TemporalGCN_6828998001460
// MI455X (gfx1250) — compile-verified, matrix operand order checked by reading, not run on hardware
//
#include <hip/hip_runtime.h>
#include <hip/hip_bf16.h>

// ---------------------------------------------------------------------------
// Problem constants (from the reference)
// ---------------------------------------------------------------------------
constexpr int B    = 128;
constexpr int C_IN = 9;
constexpr int T    = 2048;
constexpr int P1   = T / 2;      // 1024 after first pool
constexpr int R    = T / 4;      // 512  after second pool
constexpr int NN   = B * R;      // 65536 graph nodes
constexpr int HID  = 128;
constexpr int OUT  = 12;

typedef __attribute__((ext_vector_type(16))) __bf16 v16bf;
typedef __attribute__((ext_vector_type(8)))  float  v8f;

// round-to-nearest-even f32 -> bf16
__device__ __forceinline__ __bf16 f2bf(float f) {
    unsigned u = __float_as_uint(f);
    u += 0x7FFFu + ((u >> 16) & 1u);
    unsigned short h = (unsigned short)(u >> 16);
    return __builtin_bit_cast(__bf16, h);
}

// ---------------------------------------------------------------------------
// Temporal conv stage 1: conv1d(9->16, k=5, pad=2) + ReLU + maxpool2
// ---------------------------------------------------------------------------
__global__ void tconv1_kernel(const float* __restrict__ x,
                              const float* __restrict__ w,   // [16,9,5]
                              const float* __restrict__ bias,
                              float* __restrict__ out1) {
    int idx = blockIdx.x * blockDim.x + threadIdx.x;
    if (idx >= B * 16 * P1) return;
    int p  = idx % P1;
    int oc = (idx / P1) % 16;
    int bb = idx / (P1 * 16);
    float m = -__builtin_inff();
    #pragma unroll
    for (int s = 0; s < 2; ++s) {
        int t0 = 2 * p + s;
        float acc = bias[oc];
        for (int ic = 0; ic < C_IN; ++ic) {
            const float* xp = x + (bb * C_IN + ic) * T;
            const float* wp = w + (oc * C_IN + ic) * 5;
            #pragma unroll
            for (int k = 0; k < 5; ++k) {
                int t = t0 + k - 2;
                if (t >= 0 && t < T) acc = fmaf(xp[t], wp[k], acc);
            }
        }
        m = fmaxf(m, fmaxf(acc, 0.0f));
    }
    out1[idx] = m;
}

// ---------------------------------------------------------------------------
// Temporal conv stage 2: conv1d(16->32, k=5, pad=2) + ReLU + maxpool2
// writes node features directly as bf16, node-major [NN,32]
// ---------------------------------------------------------------------------
__global__ void tconv2_kernel(const float* __restrict__ h1,   // [B,16,1024]
                              const float* __restrict__ w,    // [32,16,5]
                              const float* __restrict__ bias,
                              __bf16* __restrict__ nodes) {   // [NN,32]
    int idx = blockIdx.x * blockDim.x + threadIdx.x;
    if (idx >= B * 32 * R) return;
    int r  = idx % R;
    int oc = (idx / R) % 32;
    int bb = idx / (R * 32);
    float m = -__builtin_inff();
    #pragma unroll
    for (int s = 0; s < 2; ++s) {
        int p0 = 2 * r + s;
        float acc = bias[oc];
        for (int ic = 0; ic < 16; ++ic) {
            const float* hp = h1 + (bb * 16 + ic) * P1;
            const float* wp = w + (oc * 16 + ic) * 5;
            #pragma unroll
            for (int k = 0; k < 5; ++k) {
                int p = p0 + k - 2;
                if (p >= 0 && p < P1) acc = fmaf(hp[p], wp[k], acc);
            }
        }
        m = fmaxf(m, fmaxf(acc, 0.0f));
    }
    nodes[(bb * R + r) * 32 + oc] = f2bf(m);
}

// ---------------------------------------------------------------------------
// Weight pre-swizzle: f32 W[KIN,HID] -> bf16 in per-lane WMMA B layout:
//   Wsw[((chunk*8 + colTile)*32 + lane)*16 + i]
// where lane owns column colTile*16 + (lane&15), kbase = (lane>>4)*8,
//   i<8  -> K = chunk*32 + kbase + i
//   i>=8 -> K = chunk*32 + 16 + kbase + (i-8)
// Each lane's 16 bf16 fragment becomes 32 contiguous bytes.
// ---------------------------------------------------------------------------
__global__ void wswizzle_kernel(const float* __restrict__ W,  // [KIN, HID] f32
                                __bf16* __restrict__ Wsw, int KIN) {
    int idx = blockIdx.x * blockDim.x + threadIdx.x;
    if (idx >= KIN * HID) return;
    int i     = idx & 15;
    int lane  = (idx >> 4) & 31;
    int ct    = (idx >> 9) & 7;
    int chunk = idx >> 12;
    int kbase = (lane >> 4) * 8;
    int kk    = (i < 8) ? (kbase + i) : (16 + kbase + (i - 8));
    int k     = chunk * 32 + kk;
    int n     = ct * 16 + (lane & 15);
    Wsw[idx] = f2bf(W[k * HID + n]);
}

// ---------------------------------------------------------------------------
// small utilities
// ---------------------------------------------------------------------------
__global__ void fill_f32_kernel(float* __restrict__ p, float v, int n) {
    int i = blockIdx.x * blockDim.x + threadIdx.x;
    if (i < n) p[i] = v;
}

__global__ void deg_count_kernel(const int* __restrict__ ei, int E,
                                 float* __restrict__ deg) {
    int e = blockIdx.x * blockDim.x + threadIdx.x;
    if (e < E) atomicAdd(&deg[ei[E + e]], 1.0f);   // col = target
}

__global__ void deg_rsqrt_kernel(float* __restrict__ deg, int n) {
    int i = blockIdx.x * blockDim.x + threadIdx.x;
    if (i < n) deg[i] = rsqrtf(deg[i]);            // deg >= 1 (self loop)
}

// ---------------------------------------------------------------------------
// GCN linear:  H[NN,128] = X[NN,KIN](bf16) @ W[KIN,128](bf16)
// One wave owns a 16-row block: A fragments stay in VGPRs while the wave
// sweeps all 8 column tiles (8 * KIN/32 WMMAs per wave). B comes from the
// pre-swizzled weight buffer as two contiguous b128 loads per fragment.
// ---------------------------------------------------------------------------
template <int KIN>
__global__ void gcn_lin_wmma_kernel(const __bf16* __restrict__ X,
                                    const __bf16* __restrict__ Wsw,
                                    float* __restrict__ H, int nNodes) {
    constexpr int CH = KIN / 32;                       // k-chunks
    const int wave = blockIdx.x * (blockDim.x >> 5) + (threadIdx.x >> 5);
    const int lane = threadIdx.x & 31;
    if (wave * 16 >= nNodes) return;                   // wave-uniform

    const int mrow  = wave * 16 + (lane & 15);
    const int kbase = (lane >> 4) * 8;

    // ---- load all A fragments once (per ISA 16-bit A 16x32 layout) ----
    v16bf a[CH];
    const __bf16* arow = X + (size_t)mrow * KIN;
    #pragma unroll
    for (int c = 0; c < CH; ++c) {
        #pragma unroll
        for (int i = 0; i < 8; ++i) a[c][i]     = arow[c * 32 + kbase + i];
        #pragma unroll
        for (int i = 0; i < 8; ++i) a[c][8 + i] = arow[c * 32 + 16 + kbase + i];
    }

    const int rbase = wave * 16 + ((lane >> 4) * 8);   // D rows this lane-half writes

    #pragma unroll
    for (int ct = 0; ct < HID / 16; ++ct) {
        v8f acc = {};
        #pragma unroll
        for (int c = 0; c < CH; ++c) {
            const v16bf* bp = (const v16bf*)(Wsw +
                ((size_t)(c * (HID / 16) + ct) * 32 + lane) * 16);
            v16bf b = *bp;                             // 32B contiguous per lane
            acc = __builtin_amdgcn_wmma_f32_16x16x32_bf16(
                false, a[c], false, b, (short)0, acc, false, false);
        }
        const int ncol = ct * 16 + (lane & 15);
        #pragma unroll
        for (int v = 0; v < 8; ++v)
            H[(size_t)(rbase + v) * HID + ncol] = acc[v];
    }
}

// ---------------------------------------------------------------------------
// Edge scatter: G[col] += dis[row]*dis[col] * H[row]   (one wave per edge)
// ---------------------------------------------------------------------------
__global__ void gcn_agg_kernel(const int* __restrict__ ei, int E,
                               const float* __restrict__ dis,
                               const float* __restrict__ H,
                               float* __restrict__ G) {
    int wave = (blockIdx.x * blockDim.x + threadIdx.x) >> 5;
    int lane = threadIdx.x & 31;
    if (wave >= E) return;
    int row = ei[wave];
    int col = ei[E + wave];
    float norm = dis[row] * dis[col];
    const float* hr = H + (size_t)row * HID;
    float*       gc = G + (size_t)col * HID;
    #pragma unroll
    for (int i = 0; i < HID / 32; ++i)
        atomicAdd(&gc[lane + 32 * i], norm * hr[lane + 32 * i]);
}

// ---------------------------------------------------------------------------
// self-loop + bias + ReLU (+ optional bf16 copy for next layer's WMMA input)
// ---------------------------------------------------------------------------
__global__ void gcn_post_kernel(const float* __restrict__ H,
                                const float* __restrict__ dis,
                                const float* __restrict__ bias,
                                float* __restrict__ G,
                                __bf16* __restrict__ Gbf, int nNodes) {
    int idx = blockIdx.x * blockDim.x + threadIdx.x;
    if (idx >= nNodes * HID) return;
    int node = idx >> 7, f = idx & (HID - 1);
    float d = dis[node];
    float v = G[idx] + d * d * H[idx] + bias[f];
    v = fmaxf(v, 0.0f);
    G[idx] = v;
    if (Gbf) Gbf[idx] = f2bf(v);
}

// ---------------------------------------------------------------------------
// mean over R nodes per sample, then FC 128 -> 12
// ---------------------------------------------------------------------------
__global__ void mean_pool_kernel(const float* __restrict__ G,
                                 float* __restrict__ pooled) {
    int idx = blockIdx.x * blockDim.x + threadIdx.x;
    if (idx >= B * HID) return;
    int b = idx >> 7, f = idx & (HID - 1);
    const float* p = G + (size_t)(b * R) * HID + f;
    float s = 0.0f;
    for (int r = 0; r < R; ++r) s += p[(size_t)r * HID];
    pooled[idx] = s * (1.0f / R);
}

__global__ void fc_kernel(const float* __restrict__ pooled,
                          const float* __restrict__ w,    // [128,12]
                          const float* __restrict__ bias, // [12]
                          float* __restrict__ out) {
    int idx = blockIdx.x * blockDim.x + threadIdx.x;
    if (idx >= B * OUT) return;
    int b = idx / OUT, o = idx % OUT;
    float s = bias[o];
    for (int f = 0; f < HID; ++f)
        s = fmaf(pooled[b * HID + f], w[f * OUT + o], s);
    out[idx] = s;
}

// ---------------------------------------------------------------------------
// launcher
// ---------------------------------------------------------------------------
extern "C" void kernel_launch(void* const* d_in, const int* in_sizes, int n_in,
                              void* d_out, int out_size, void* d_ws, size_t ws_size,
                              hipStream_t stream) {
    const float* x       = (const float*)d_in[0];
    const int*   ei      = (const int*)  d_in[1];
    const float* conv1_w = (const float*)d_in[2];
    const float* conv1_b = (const float*)d_in[3];
    const float* conv2_w = (const float*)d_in[4];
    const float* conv2_b = (const float*)d_in[5];
    const float* gcn1_w  = (const float*)d_in[6];
    const float* gcn1_b  = (const float*)d_in[7];
    const float* gcn2_w  = (const float*)d_in[8];
    const float* gcn2_b  = (const float*)d_in[9];
    const float* fc_w    = (const float*)d_in[10];
    const float* fc_b    = (const float*)d_in[11];
    float* out = (float*)d_out;

    const int E = in_sizes[1] / 2;   // edges (B * E_SINGLE)

    // ---- workspace carve-up (all 256B aligned) ----
    char* ws = (char*)d_ws;
    size_t off = 0;
    auto carve = [&](size_t bytes) {
        char* p = ws + off;
        off += (bytes + 255) & ~(size_t)255;
        return p;
    };
    __bf16* nodes_bf = (__bf16*)carve((size_t)NN * 32 * 2);
    __bf16* w1_sw    = (__bf16*)carve((size_t)32 * HID * 2);
    __bf16* w2_sw    = (__bf16*)carve((size_t)HID * HID * 2);
    float*  dis      = (float*) carve((size_t)NN * 4);
    float*  Hbuf     = (float*) carve((size_t)NN * HID * 4);
    float*  Gbuf     = (float*) carve((size_t)NN * HID * 4);
    __bf16* Gbf      = (__bf16*)carve((size_t)NN * HID * 2);
    float*  pooled   = (float*) carve((size_t)B * HID * 4);
    float*  h1       = (float*) carve((size_t)B * 16 * P1 * 4);
    (void)ws_size;

    const int TB = 256;
    auto blocks = [&](long n) { return (int)((n + TB - 1) / TB); };

    // ---- temporal conv stack ----
    tconv1_kernel<<<blocks((long)B * 16 * P1), TB, 0, stream>>>(x, conv1_w, conv1_b, h1);
    tconv2_kernel<<<blocks((long)B * 32 * R), TB, 0, stream>>>(h1, conv2_w, conv2_b, nodes_bf);

    // ---- weight pre-swizzle into WMMA B-fragment order (bf16) ----
    wswizzle_kernel<<<blocks(32 * HID), TB, 0, stream>>>(gcn1_w, w1_sw, 32);
    wswizzle_kernel<<<blocks(HID * HID), TB, 0, stream>>>(gcn2_w, w2_sw, HID);

    // ---- degrees: init to 1 (self loop), count edges, rsqrt ----
    fill_f32_kernel<<<blocks(NN), TB, 0, stream>>>(dis, 1.0f, NN);
    deg_count_kernel<<<blocks(E), TB, 0, stream>>>(ei, E, dis);
    deg_rsqrt_kernel<<<blocks(NN), TB, 0, stream>>>(dis, NN);

    const int linBlocks = (NN / 16) / (TB / 32);   // one wave per 16-row block
    const long nElems = (long)NN * HID;

    // ---- GCN layer 1 ----
    gcn_lin_wmma_kernel<32><<<linBlocks, TB, 0, stream>>>(nodes_bf, w1_sw, Hbuf, NN);
    fill_f32_kernel<<<blocks(nElems), TB, 0, stream>>>(Gbuf, 0.0f, (int)nElems);
    gcn_agg_kernel<<<blocks((long)E * 32), TB, 0, stream>>>(ei, E, dis, Hbuf, Gbuf);
    gcn_post_kernel<<<blocks(nElems), TB, 0, stream>>>(Hbuf, dis, gcn1_b, Gbuf, Gbf, NN);

    // ---- GCN layer 2 ----
    gcn_lin_wmma_kernel<HID><<<linBlocks, TB, 0, stream>>>(Gbf, w2_sw, Hbuf, NN);
    fill_f32_kernel<<<blocks(nElems), TB, 0, stream>>>(Gbuf, 0.0f, (int)nElems);
    gcn_agg_kernel<<<blocks((long)E * 32), TB, 0, stream>>>(ei, E, dis, Hbuf, Gbuf);
    gcn_post_kernel<<<blocks(nElems), TB, 0, stream>>>(Hbuf, dis, gcn2_b, Gbuf, (__bf16*)nullptr, NN);

    // ---- readout ----
    mean_pool_kernel<<<blocks(B * HID), TB, 0, stream>>>(Gbuf, pooled);
    fc_kernel<<<blocks(B * OUT), TB, 0, stream>>>(pooled, fc_w, fc_b, out);
}
